// GraphTransformer_56736517980960
// MI455X (gfx1250) — compile-verified, matrix operand order checked by reading, not run on hardware
//
#include <hip/hip_runtime.h>
#include <hip/hip_bf16.h>
#include <math.h>

typedef __attribute__((ext_vector_type(16))) _Float16 v16h;
typedef __attribute__((ext_vector_type(8)))  _Float16 v8h;
typedef __attribute__((ext_vector_type(8)))  float    v8f;
typedef __attribute__((ext_vector_type(4)))  float    v4f;

constexpr int N_NODES = 50000;
constexpr int N_EDGES = 600000;
constexpr int N_GRAPH = 1000;
constexpr int HID = 128, HEADS = 8, DH = 16, LAYERS = 3;
constexpr int FFN = 4 * HID;           // 512
constexpr float EPS = 1e-5f;

static __device__ __forceinline__ float gelu_exact(float x) {
  return 0.5f * x * (1.0f + erff(x * 0.70710678118654752f));
}
static __device__ __forceinline__ float wave_sum(float v) {
#pragma unroll
  for (int m = 16; m > 0; m >>= 1) v += __shfl_xor(v, m, 32);
  return v;
}
static __device__ __forceinline__ void atomicMaxF(float* addr, float val) {
  int* ia = (int*)addr;
  int cur = __float_as_int(*addr);
  while (__int_as_float(cur) < val) {
    int prev = atomicCAS(ia, cur, __float_as_int(val));
    if (prev == cur) break;
    cur = prev;
  }
}

// ---------------------------------------------------------------------------
// Weight pre-swizzle: W[K][Nc] f32 row-major -> f16 WMMA-B fragment-major.
// Fragment (nt, kt): 32 lanes x 16 halves contiguous; lane n = nt*16+(lane&15),
// k base = kt*32 + (lane>=16 ? 8 : 0); element pairs follow ISA 7.12.2.
// One 32B chunk per lane -> two global_load_b128 per fragment.
// ---------------------------------------------------------------------------
__global__ void pack_b_kernel(const float* __restrict__ W,
                              _Float16* __restrict__ P, int K, int Nc) {
  const int i = blockIdx.x * 256 + threadIdx.x;
  if (i >= K * Nc) return;
  const int idx  = i & 15;
  const int lane = (i >> 4) & 31;
  const int t    = i >> 9;            // fragment id = nt*ktiles + kt
  const int ktiles = K >> 5;
  const int kt = t % ktiles;
  const int nt = t / ktiles;
  const int n  = (nt << 4) + (lane & 15);
  const int kb = (lane >> 4) << 3;
  const int g  = idx >> 1;
  const int kk = (kt << 5) + kb +
                 ((g < 4) ? (g << 1) : (16 + ((g - 4) << 1))) + (idx & 1);
  P[i] = (_Float16)W[(size_t)kk * Nc + n];
}

// ---------------------------------------------------------------------------
// WMMA GEMM: C[M,Nc] = A[M,K] @ B + bias.  A f32/f16 global -> f16 LDS tile.
// B pre-swizzled fragment-major f16.  Block = 256 thr = 8 wave32s.
// Block tile: 64 rows x (8 waves * 16 cols); each wave: 1 col-tile,
// 4 row-subtiles -> 4 WMMA per B-fragment load.  Col-tile guard is forced
// scalar (readfirstlane) so WMMA runs with EXEC all-ones.  B fragment is
// issued before the barrier to overlap with A staging; all 4 A fragments
// are loaded before the 4 back-to-back WMMAs (one dscnt wait).
// ---------------------------------------------------------------------------
template <bool A_HALF, bool W32, bool W16>
__global__ __launch_bounds__(256) void wmma_gemm_kernel(
    const void* __restrict__ Av, const _Float16* __restrict__ B,
    const float* __restrict__ bias, float* __restrict__ C,
    _Float16* __restrict__ C16, int M, int K, int Nc) {
  __shared__ _Float16 sA[64][40];            // 64 rows x 32 k (+pad), 5KB
  const int wave = threadIdx.x >> 5;
  const int lane = threadIdx.x & 31;
  const int row0 = blockIdx.x << 6;
  const int ntiles = Nc >> 4;
  const int ktiles = K >> 5;
  // wave-uniform by construction; readfirstlane makes it an SGPR -> s_cbranch
  const int tile = __builtin_amdgcn_readfirstlane((int)(blockIdx.y << 3) + wave);
  const bool active = tile < ntiles;

  const _Float16* Ah = (const _Float16*)Av;
  const float*    Af = (const float*)Av;

  v8f acc[4];
#pragma unroll
  for (int s = 0; s < 4; ++s) acc[s] = (v8f)(0.0f);

  const int sr = threadIdx.x >> 2;            // staging row 0..63
  const int sc = (threadIdx.x & 3) << 3;      // staging col 0,8,16,24
  const int m  = lane & 15;
  const int kb = (lane >> 4) << 3;

  for (int k0 = 0; k0 < K; k0 += 32) {
    // ---- stage A tile: each thread 8 halves (16B vector path) ----
    {
      int row = row0 + sr;
      if (row >= M) row = M - 1;              // clamp (M%16==0 covers stores)
      const size_t abase = (size_t)row * K + k0 + sc;
      if (A_HALF) {
        *(v8h*)&sA[sr][sc] = *(const v8h*)(Ah + abase);
      } else {
        const v4f f0 = *(const v4f*)(Af + abase);
        const v4f f1 = *(const v4f*)(Af + abase + 4);
        v8h hv;
#pragma unroll
        for (int j = 0; j < 4; ++j) {
          hv[j]     = (_Float16)f0[j];
          hv[j + 4] = (_Float16)f1[j];
        }
        *(v8h*)&sA[sr][sc] = hv;
      }
      if (k0 + 32 < K) {                      // gfx1250 global_prefetch_b8
        if (A_HALF) __builtin_prefetch(Ah + abase + 32, 0, 1);
        else        __builtin_prefetch(Af + abase + 32, 0, 1);
      }
    }

    // ---- B fragment issued pre-barrier: overlaps barrier + LDS drain ----
    v16h b;
    if (active) {
      const size_t boff = ((size_t)tile * ktiles + (k0 >> 5)) * 32 + lane;
      b = ((const v16h*)B)[boff];
    }
    __syncthreads();

    if (active) {
      // all 4 A fragments first (8x ds_load_b128, single wait) ...
      v16h afr[4];
#pragma unroll
      for (int s = 0; s < 4; ++s) {
        const v8h lo = *(const v8h*)&sA[s * 16 + m][kb];
        const v8h hi = *(const v8h*)&sA[s * 16 + m][kb + 16];
        afr[s] = __builtin_shufflevector(lo, hi, 0, 1, 2, 3, 4, 5, 6, 7, 8, 9,
                                         10, 11, 12, 13, 14, 15);
      }
      // ... then 4 back-to-back WMMAs (independent accumulators, no hazard)
#pragma unroll
      for (int s = 0; s < 4; ++s)
        acc[s] = __builtin_amdgcn_wmma_f32_16x16x32_f16(
            false, afr[s], false, b, (short)0, acc[s], false, false);
    }
    __syncthreads();
  }

  if (active) {
    const int n  = (tile << 4) + (lane & 15);
    const float bv = bias ? bias[n] : 0.0f;
#pragma unroll
    for (int s = 0; s < 4; ++s) {
      if (row0 + s * 16 >= M) break;          // M%16==0 -> subtile guard
      const int mb = row0 + s * 16 + ((lane >> 4) << 3);
#pragma unroll
      for (int r = 0; r < 8; ++r) {
        const float v = acc[s][r] + bv;
        const size_t idx = (size_t)(mb + r) * Nc + n;
        if (W32) C[idx]   = v;
        if (W16) C16[idx] = (_Float16)v;
      }
    }
  }
}

// ---------------------------------------------------------------------------
// Elementwise / row-wise kernels (wave32-per-row LayerNorm reductions)
// ---------------------------------------------------------------------------
__global__ void starts_kernel(const long long* __restrict__ batch,
                              int* __restrict__ starts, int n) {
  const int i = blockIdx.x * 256 + threadIdx.x;
  if (i >= n) return;
  if (i == 0 || batch[i] != batch[i - 1]) starts[(int)batch[i]] = i;
}

__global__ void pos_kernel(const long long* __restrict__ batch,
                           const int* __restrict__ starts,
                           int* __restrict__ pos, int n) {
  const int i = blockIdx.x * 256 + threadIdx.x;
  if (i < n) pos[i] = i - starts[(int)batch[i]];
}

// y = gelu(LN(in)*g+b) (+ pe[pos[row]] if pe); one wave per row, Cw in {64,128}
__global__ __launch_bounds__(256) void ln_act_kernel(
    const float* __restrict__ in, const float* __restrict__ g,
    const float* __restrict__ b, const float* __restrict__ pe,
    const int* __restrict__ pos, float* __restrict__ out32,
    _Float16* __restrict__ out16, int Cw, int M) {
  const int wave = threadIdx.x >> 5, lane = threadIdx.x & 31;
  const int row = blockIdx.x * 8 + wave;
  if (row >= M) return;
  const int per = Cw >> 5;
  const float* rp = in + (size_t)row * Cw;
  float x[4], s = 0.f, s2 = 0.f;
  for (int j = 0; j < per; ++j) {
    x[j] = rp[lane * per + j];
    s += x[j]; s2 += x[j] * x[j];
  }
  s = wave_sum(s); s2 = wave_sum(s2);
  const float mean = s / Cw;
  const float inv  = rsqrtf(s2 / Cw - mean * mean + EPS);
  const float* perow = nullptr;
  if (pe) { int p = pos[row]; if (p > 99) p = 99; perow = pe + (size_t)p * Cw; }
  for (int j = 0; j < per; ++j) {
    const int c = lane * per + j;
    float y = (x[j] - mean) * inv * g[c] + b[c];
    y = gelu_exact(y);
    if (perow) y += perow[c];
    const size_t idx = (size_t)row * Cw + c;
    if (out32) out32[idx] = y;
    if (out16) out16[idx] = (_Float16)y;
  }
}

__global__ void attn_init_kernel(float* __restrict__ attnout,
                                 float* __restrict__ mmax,
                                 float* __restrict__ den) {
  const int i = blockIdx.x * 256 + threadIdx.x;
  if (i < N_NODES * HID) attnout[i] = 0.f;
  if (i < N_NODES * HEADS) { mmax[i] = -3.0e38f; den[i] = 0.f; }
}

// pass 1: logits per (edge, head) + segment max over dst (32B vector loads)
__global__ __launch_bounds__(256) void edge_logits_kernel(
    const long long* __restrict__ ei, const _Float16* __restrict__ q16,
    const _Float16* __restrict__ k16, const _Float16* __restrict__ ep16,
    float* __restrict__ logits, float* __restrict__ mmax) {
  const int t = blockIdx.x * 256 + threadIdx.x;
  if (t >= N_EDGES * HEADS) return;
  const int e = t >> 3, hd = t & 7;
  const int src = (int)ei[e];
  const int dst = (int)ei[N_EDGES + e];
  const v16h qq = *(const v16h*)(q16 + (size_t)dst * HID + hd * DH);
  const v16h kk = *(const v16h*)(k16 + (size_t)src * HID + hd * DH);
  const v16h pp = *(const v16h*)(ep16 + (size_t)e * HID + hd * DH);
  float acc = 0.f;
#pragma unroll
  for (int j = 0; j < DH; ++j)
    acc += (float)qq[j] * ((float)kk[j] + (float)pp[j]);
  acc *= 0.25f;                               // 1/sqrt(DH)
  logits[t] = acc;
  atomicMaxF(&mmax[(size_t)dst * HEADS + hd], acc);
}

// pass 2: ex = exp(l - m[dst]); den += ex; attnout += ex * (v[src]+e_proj)
__global__ __launch_bounds__(256) void edge_accum_kernel(
    const long long* __restrict__ ei, const _Float16* __restrict__ v16,
    const _Float16* __restrict__ ep16, const float* __restrict__ logits,
    const float* __restrict__ mmax, float* __restrict__ den,
    float* __restrict__ attnout) {
  const int t = blockIdx.x * 256 + threadIdx.x;
  if (t >= N_EDGES * HEADS) return;
  const int e = t >> 3, hd = t & 7;
  const int src = (int)ei[e];
  const int dst = (int)ei[N_EDGES + e];
  const float ex = __expf(logits[t] - mmax[(size_t)dst * HEADS + hd]);
  atomicAdd(&den[(size_t)dst * HEADS + hd], ex);
  const v16h vv = *(const v16h*)(v16 + (size_t)src * HID + hd * DH);
  const v16h pp = *(const v16h*)(ep16 + (size_t)e * HID + hd * DH);
  float* op = attnout + (size_t)dst * HID + hd * DH;
#pragma unroll
  for (int j = 0; j < DH; ++j)
    atomicAdd(&op[j], ex * ((float)vv[j] + (float)pp[j]));
}

// normalize, beta-gate, residual + LN1; one wave per node
__global__ __launch_bounds__(256) void combine_kernel(
    const float* __restrict__ attnout, const float* __restrict__ den,
    const float* __restrict__ skip, const float* __restrict__ Wbeta,
    const float* __restrict__ g, const float* __restrict__ bb,
    float* __restrict__ h, _Float16* __restrict__ h16) {
  const int wave = threadIdx.x >> 5, lane = threadIdx.x & 31;
  const int node = blockIdx.x * 8 + wave;
  if (node >= N_NODES) return;
  float o[4], sk[4], dot = 0.f;
#pragma unroll
  for (int j = 0; j < 4; ++j) {
    const int c = lane * 4 + j;
    const float d = den[(size_t)node * HEADS + (c >> 4)];
    o[j]  = attnout[(size_t)node * HID + c] / d;
    sk[j] = skip[(size_t)node * HID + c];
    dot += Wbeta[c] * o[j] + Wbeta[HID + c] * sk[j] +
           Wbeta[2 * HID + c] * (o[j] - sk[j]);
  }
  dot = wave_sum(dot);
  const float beta = 1.f / (1.f + __expf(-dot));
  float t[4], s = 0.f, s2 = 0.f;
#pragma unroll
  for (int j = 0; j < 4; ++j) {
    const int c = lane * 4 + j;
    const float a = beta * sk[j] + (1.f - beta) * o[j];
    t[j] = h[(size_t)node * HID + c] + a;
    s += t[j]; s2 += t[j] * t[j];
  }
  s = wave_sum(s); s2 = wave_sum(s2);
  const float mean = s / HID;
  const float inv  = rsqrtf(s2 / HID - mean * mean + EPS);
#pragma unroll
  for (int j = 0; j < 4; ++j) {
    const int c = lane * 4 + j;
    const float y = (t[j] - mean) * inv * g[c] + bb[c];
    h[(size_t)node * HID + c]   = y;
    h16[(size_t)node * HID + c] = (_Float16)y;
  }
}

// h = LN(h + f); one wave per row
__global__ __launch_bounds__(256) void resid_ln_kernel(
    const float* __restrict__ f, const float* __restrict__ g,
    const float* __restrict__ bb, float* __restrict__ h,
    _Float16* __restrict__ h16) {
  const int wave = threadIdx.x >> 5, lane = threadIdx.x & 31;
  const int node = blockIdx.x * 8 + wave;
  if (node >= N_NODES) return;
  float t[4], s = 0.f, s2 = 0.f;
#pragma unroll
  for (int j = 0; j < 4; ++j) {
    const int c = lane * 4 + j;
    t[j] = h[(size_t)node * HID + c] + f[(size_t)node * HID + c];
    s += t[j]; s2 += t[j] * t[j];
  }
  s = wave_sum(s); s2 = wave_sum(s2);
  const float mean = s / HID;
  const float inv  = rsqrtf(s2 / HID - mean * mean + EPS);
#pragma unroll
  for (int j = 0; j < 4; ++j) {
    const int c = lane * 4 + j;
    const float y = (t[j] - mean) * inv * g[c] + bb[c];
    h[(size_t)node * HID + c]   = y;
    h16[(size_t)node * HID + c] = (_Float16)y;
  }
}

__global__ void gelu_f16_kernel(const float* __restrict__ in,
                                _Float16* __restrict__ out, int n) {
  const int i = blockIdx.x * 256 + threadIdx.x;
  if (i < n) out[i] = (_Float16)gelu_exact(in[i]);
}

// ---------------------------------------------------------------------------
// Host launcher
// ---------------------------------------------------------------------------
extern "C" void kernel_launch(void* const* d_in, const int* in_sizes, int n_in,
                              void* d_out, int out_size, void* d_ws,
                              size_t ws_size, hipStream_t stream) {
  (void)in_sizes; (void)n_in; (void)out_size; (void)ws_size;

  // --- inputs, flattened in setup_inputs() dict insertion order ---
  const float*     x     = (const float*)d_in[0];
  const long long* ei    = (const long long*)d_in[1];
  const float*     ea    = (const float*)d_in[2];
  const long long* batch = (const long long*)d_in[3];
  const float* ne_W  = (const float*)d_in[4];
  const float* ne_b  = (const float*)d_in[5];
  const float* ne_g  = (const float*)d_in[6];
  const float* ne_be = (const float*)d_in[7];
  const float* ee_W  = (const float*)d_in[8];
  const float* ee_b  = (const float*)d_in[9];
  const float* ee_g  = (const float*)d_in[10];
  const float* ee_be = (const float*)d_in[11];
  const float* op_W1 = (const float*)d_in[12];
  const float* op_b1 = (const float*)d_in[13];
  const float* op_g  = (const float*)d_in[14];
  const float* op_be = (const float*)d_in[15];
  const float* op_W2 = (const float*)d_in[16];
  const float* op_b2 = (const float*)d_in[17];
  const int LP = 18;
  auto lw = [&](int l, int k) { return (const float*)d_in[18 + l * LP + k]; };
  // 0 Wq 1 bq 2 Wk 3 bk 4 Wv 5 bv 6 We 7 Wskip 8 bskip 9 Wbeta
  // 10 ln1_g 11 ln1_b 12 ln2_g 13 ln2_b 14 ffn_W1 15 ffn_b1 16 ffn_W2 17 ffn_b2
  const float* pe = (const float*)d_in[18 + LAYERS * LP];   // index 72

  float* out = (float*)d_out;

  // --- workspace bump allocator ---
  char* base = (char*)d_ws;
  size_t off = 0;
  auto alloc = [&](size_t bytes) -> void* {
    off = (off + 255) & ~(size_t)255;
    void* p = base + off;
    off += bytes;
    return p;
  };

  int*       starts  = (int*)alloc((size_t)N_GRAPH * 4);
  int*       pos     = (int*)alloc((size_t)N_NODES * 4);
  float*     h       = (float*)alloc((size_t)N_NODES * HID * 4);
  _Float16*  h16     = (_Float16*)alloc((size_t)N_NODES * HID * 2);
  _Float16*  e16     = (_Float16*)alloc((size_t)N_EDGES * HID * 2);
  _Float16*  ep16    = (_Float16*)alloc((size_t)N_EDGES * HID * 2);
  _Float16*  q16     = (_Float16*)alloc((size_t)N_NODES * HID * 2);
  _Float16*  k16     = (_Float16*)alloc((size_t)N_NODES * HID * 2);
  _Float16*  v16     = (_Float16*)alloc((size_t)N_NODES * HID * 2);
  float*     skipb   = (float*)alloc((size_t)N_NODES * HID * 4);
  float*     attnout = (float*)alloc((size_t)N_NODES * HID * 4);
  float*     mmax    = (float*)alloc((size_t)N_NODES * HEADS * 4);
  float*     den     = (float*)alloc((size_t)N_NODES * HEADS * 4);
  float*     logits  = (float*)alloc((size_t)N_EDGES * HEADS * 4);
  _Float16*  t16     = (_Float16*)alloc((size_t)N_NODES * FFN * 2);
  _Float16*  o16     = (_Float16*)alloc((size_t)N_NODES * 64 * 2);
  float*     big     = (float*)alloc((size_t)N_EDGES * HID * 4);  // >= N*FFN

  // f16 fragment-major weight copies
  _Float16* wne  = (_Float16*)alloc(64 * 128 * 2);
  _Float16* wee  = (_Float16*)alloc(32 * 128 * 2);
  _Float16* wop1 = (_Float16*)alloc(128 * 64 * 2);
  _Float16* wop2 = (_Float16*)alloc(64 * 32 * 2);
  _Float16 *wq[LAYERS], *wk[LAYERS], *wv[LAYERS], *we[LAYERS], *wsk[LAYERS],
           *wf1[LAYERS], *wf2[LAYERS];
  for (int l = 0; l < LAYERS; ++l) {
    wq[l]  = (_Float16*)alloc(128 * 128 * 2);
    wk[l]  = (_Float16*)alloc(128 * 128 * 2);
    wv[l]  = (_Float16*)alloc(128 * 128 * 2);
    we[l]  = (_Float16*)alloc(128 * 128 * 2);
    wsk[l] = (_Float16*)alloc(128 * 128 * 2);
    wf1[l] = (_Float16*)alloc(128 * 512 * 2);
    wf2[l] = (_Float16*)alloc(512 * 128 * 2);
  }

  auto pack = [&](const float* s, _Float16* d, int K, int Nc) {
    pack_b_kernel<<<(K * Nc + 255) / 256, 256, 0, stream>>>(s, d, K, Nc);
  };
  pack(ne_W, wne, 64, 128);
  pack(ee_W, wee, 32, 128);
  pack(op_W1, wop1, 128, 64);
  pack(op_W2, wop2, 64, 32);
  for (int l = 0; l < LAYERS; ++l) {
    pack(lw(l, 0), wq[l], 128, 128);
    pack(lw(l, 2), wk[l], 128, 128);
    pack(lw(l, 4), wv[l], 128, 128);
    pack(lw(l, 6), we[l], 128, 128);
    pack(lw(l, 7), wsk[l], 128, 128);
    pack(lw(l, 14), wf1[l], 128, 512);
    pack(lw(l, 16), wf2[l], 512, 128);
  }

  auto grid_of = [](int M, int Nc) {
    return dim3((M + 63) / 64, ((Nc / 16) + 7) / 8);
  };
  auto gemm_f32 = [&](const float* A, const _Float16* B, const float* bias,
                      float* C, int M, int K, int Nc) {
    wmma_gemm_kernel<false, true, false><<<grid_of(M, Nc), 256, 0, stream>>>(
        A, B, bias, C, nullptr, M, K, Nc);
  };
  auto gemm_h16 = [&](const _Float16* A, const _Float16* B, const float* bias,
                      _Float16* C16, int M, int K, int Nc) {
    wmma_gemm_kernel<true, false, true><<<grid_of(M, Nc), 256, 0, stream>>>(
        A, B, bias, nullptr, C16, M, K, Nc);
  };
  auto gemm_h32 = [&](const _Float16* A, const _Float16* B, const float* bias,
                      float* C, int M, int K, int Nc) {
    wmma_gemm_kernel<true, true, false><<<grid_of(M, Nc), 256, 0, stream>>>(
        A, B, bias, C, nullptr, M, K, Nc);
  };

  const int nb_nodes  = (N_NODES + 255) / 256;
  const int row_blk_n = N_NODES / 8;
  const int row_blk_e = N_EDGES / 8;

  starts_kernel<<<nb_nodes, 256, 0, stream>>>(batch, starts, N_NODES);
  pos_kernel<<<nb_nodes, 256, 0, stream>>>(batch, starts, pos, N_NODES);

  // node encoder: h = gelu(LN(x@W+b)) + pe[pos]
  gemm_f32(x, wne, ne_b, big, N_NODES, 64, HID);
  ln_act_kernel<<<row_blk_n, 256, 0, stream>>>(big, ne_g, ne_be, pe, pos, h,
                                               h16, HID, N_NODES);
  // edge encoder (f16 only)
  gemm_f32(ea, wee, ee_b, big, N_EDGES, 32, HID);
  ln_act_kernel<<<row_blk_e, 256, 0, stream>>>(big, ee_g, ee_be, nullptr,
                                               nullptr, nullptr, e16, HID,
                                               N_EDGES);

  for (int l = 0; l < LAYERS; ++l) {
    gemm_h16(e16, we[l], nullptr, ep16, N_EDGES, HID, HID);
    gemm_h16(h16, wq[l], lw(l, 1), q16, N_NODES, HID, HID);
    gemm_h16(h16, wk[l], lw(l, 3), k16, N_NODES, HID, HID);
    gemm_h16(h16, wv[l], lw(l, 5), v16, N_NODES, HID, HID);
    gemm_h32(h16, wsk[l], lw(l, 8), skipb, N_NODES, HID, HID);

    attn_init_kernel<<<(N_NODES * HID + 255) / 256, 256, 0, stream>>>(attnout,
                                                                      mmax, den);
    const int eb = (N_EDGES * HEADS + 255) / 256;
    edge_logits_kernel<<<eb, 256, 0, stream>>>(ei, q16, k16, ep16, logits, mmax);
    edge_accum_kernel<<<eb, 256, 0, stream>>>(ei, v16, ep16, logits, mmax, den,
                                              attnout);
    combine_kernel<<<row_blk_n, 256, 0, stream>>>(attnout, den, skipb, lw(l, 9),
                                                  lw(l, 10), lw(l, 11), h, h16);
    gemm_h32(h16, wf1[l], lw(l, 15), big, N_NODES, HID, FFN);
    gelu_f16_kernel<<<(N_NODES * FFN + 255) / 256, 256, 0, stream>>>(
        big, t16, N_NODES * FFN);
    gemm_h32(t16, wf2[l], lw(l, 17), skipb, N_NODES, FFN, HID);
    resid_ln_kernel<<<row_blk_n, 256, 0, stream>>>(skipb, lw(l, 12), lw(l, 13),
                                                   h, h16);
  }

  gemm_h32(h16, wop1, op_b1, big, N_NODES, HID, 64);
  ln_act_kernel<<<row_blk_n, 256, 0, stream>>>(big, op_g, op_be, nullptr,
                                               nullptr, nullptr, o16, 64,
                                               N_NODES);
  gemm_h32(o16, wop2, op_b2, out, N_NODES, 64, 32);
}
